// PyTorchMoE_FC_54211077210523
// MI455X (gfx1250) — matrix operand order repaired and checked by reading, not compiled, not run on hardware
//
#include <hip/hip_runtime.h>

// ---------------------------------------------------------------------------
// MoE FFN (top-1, E=2) for MI455X / gfx1250.
//   out[t,:] = x[t,:] @ We[e*] + be[e*],  e* = argmax(x[t]@Wg + bg)
// Pipeline:
//   1) cvt_x   : x  f32 -> bf16                (ws)   [one-time]
//   2) cvt_we  : We f32 -> bf16, TRANSPOSED [e][h][c] (ws)   [one-time]
//   3) gate    : routing lists via atomics
//   4) gemm    : TDM gather-load A (16 token rows), TDM 2D tile B (double
//                buffered), v_wmma_f32_16x16x32_bf16 x2 per K-step.
// ---------------------------------------------------------------------------

typedef __attribute__((ext_vector_type(16))) __bf16 v16bf;
typedef __attribute__((ext_vector_type(8)))  __bf16 v8bf;
typedef __attribute__((ext_vector_type(8)))  float  v8f;
typedef __attribute__((ext_vector_type(4)))  unsigned u32x4;
typedef __attribute__((ext_vector_type(8)))  int      i32x8;
typedef __attribute__((ext_vector_type(4)))  int      i32x4;

#define D_MODEL    1024
#define D_HIDDEN   4096
#define NUM_EXPERT 2
#define T_TOKENS   16384
#define TILE_M     16
#define BN         128            // N columns per block (8 waves x 16)
#define KSTEP      64             // K per main-loop iter (2 WMMAs)
#define NITER      (D_MODEL / KSTEP)
#define MTILES_PER_EXPERT (T_TOKENS / TILE_M)   // 1024

// LDS layouts produced by TDM padding:
//  A: 1024 bf16/row (2048B) + 16B pad per 1KB  -> row stride 2080B
#define A_ROW_BYTES 2080
//  B: 64 bf16/row (128B) + 32B pad per row     -> row stride 160B
#define B_ROW_BYTES 160
#define B_BUF_BYTES (BN * B_ROW_BYTES)          // 20480B per buffer

// workspace layout (bytes)
#define WS_IDX_OFF  64
#define WS_XBF_OFF  (1u << 20)                          // 1MB
#define WS_WET_OFF  (WS_XBF_OFF + (size_t)T_TOKENS * D_MODEL * 2)  // +32MB

__device__ __forceinline__ __bf16 f2bf(float f) {
  unsigned u = __builtin_bit_cast(unsigned, f);
  unsigned r = u + 0x7FFFu + ((u >> 16) & 1u);
  unsigned short h = (unsigned short)(r >> 16);
  return __builtin_bit_cast(__bf16, h);
}

__device__ __forceinline__ unsigned lds_off(const void* p) {
  // LDS aperture: low 32 bits of a generic shared-space address are the LDS
  // byte offset (ISA 10.2 aperture mapping).
  return (unsigned)(uintptr_t)p;
}

// ---------------------------------------------------------------------------
__global__ void moe_init(int* __restrict__ counts) {
  if (threadIdx.x < NUM_EXPERT) counts[threadIdx.x] = 0;
}

// x f32 -> bf16 flat copy (8 elems / thread)
__global__ void __launch_bounds__(256)
moe_cvt_x(const float* __restrict__ x, __bf16* __restrict__ xbf) {
  size_t i = ((size_t)blockIdx.x * blockDim.x + threadIdx.x) * 8;
  float4 a = *(const float4*)(x + i);
  float4 b = *(const float4*)(x + i + 4);
  v8bf v;
  v[0] = f2bf(a.x); v[1] = f2bf(a.y); v[2] = f2bf(a.z); v[3] = f2bf(a.w);
  v[4] = f2bf(b.x); v[5] = f2bf(b.y); v[6] = f2bf(b.z); v[7] = f2bf(b.w);
  *(v8bf*)(xbf + i) = v;
}

// We [e][c][h] f32 -> WeT [e][h][c] bf16 via 32x32 LDS tile transpose
__global__ void __launch_bounds__(256)
moe_cvt_we(const float* __restrict__ We, __bf16* __restrict__ WeT) {
  __shared__ float tile[32][33];
  int eb = blockIdx.z;
  int c0 = blockIdx.x * 32;
  int h0 = blockIdx.y * 32;
  int tx = threadIdx.x & 31, ty = threadIdx.x >> 5;          // 32 x 8
  const float* src = We + (size_t)eb * D_MODEL * D_HIDDEN;
  for (int i = 0; i < 32; i += 8)
    tile[ty + i][tx] = src[(size_t)(c0 + ty + i) * D_HIDDEN + h0 + tx];
  __syncthreads();
  __bf16* dst = WeT + (size_t)eb * D_MODEL * D_HIDDEN;
  for (int i = 0; i < 32; i += 8)
    dst[(size_t)(h0 + ty + i) * D_MODEL + c0 + tx] = f2bf(tile[tx][ty + i]);
}

// ---------------------------------------------------------------------------
__global__ void __launch_bounds__(256)
moe_gate(const float* __restrict__ x, const float* __restrict__ Wg,
         const float* __restrict__ bg, int* __restrict__ counts,
         int* __restrict__ idx) {
  int t = blockIdx.x * blockDim.x + threadIdx.x;
  if (t >= T_TOKENS) return;
  const float* xr = x + (size_t)t * D_MODEL;
  float l0 = bg[0], l1 = bg[1];
  for (int c = 0; c < D_MODEL; c += 4) {
    float4 xv = *(const float4*)(xr + c);
    float4 w0 = *(const float4*)(Wg + c * 2);
    float4 w1 = *(const float4*)(Wg + c * 2 + 4);
    l0 = fmaf(xv.x, w0.x, l0); l1 = fmaf(xv.x, w0.y, l1);
    l0 = fmaf(xv.y, w0.z, l0); l1 = fmaf(xv.y, w0.w, l1);
    l0 = fmaf(xv.z, w1.x, l0); l1 = fmaf(xv.z, w1.y, l1);
    l0 = fmaf(xv.w, w1.z, l0); l1 = fmaf(xv.w, w1.w, l1);
  }
  int e = (l1 > l0) ? 1 : 0;                 // first-max tie-break like top_k
  int pos = atomicAdd(&counts[e], 1);
  idx[e * T_TOKENS + pos] = t;
}

// ---------------------------------------------------------------------------
// TDM descriptor issue (CDNA5 ISA). This toolchain uses
// the 6-arg builtin form: (g0 x4, g1 x8, g2 x4, g3 x4, extra x8, cpol).
__device__ __forceinline__ void tdm_issue(u32x4 g0, i32x8 g1, i32x4 g2, i32x4 g3) {
  i32x8 zx = { 0, 0, 0, 0, 0, 0, 0, 0 };
  __builtin_amdgcn_tensor_load_to_lds(g0, g1, g2, g3, zx, 0);
}

// ---------------------------------------------------------------------------
__global__ void __launch_bounds__(256)
moe_expert_gemm(const __bf16* __restrict__ xbf, const __bf16* __restrict__ WeT,
                const float* __restrict__ be, const int* __restrict__ counts,
                const int* __restrict__ idx, float* __restrict__ out) {
  __shared__ alignas(16) char Abuf[TILE_M * A_ROW_BYTES];   // 33280B, TDM-padded
  __shared__ alignas(16) char Bbuf[2][B_BUF_BYTES];         // 2 x 20480B
  __shared__ int tok[TILE_M];

  const int e     = blockIdx.x >> 10;
  const int tileM = blockIdx.x & (MTILES_PER_EXPERT - 1);
  const int n0blk = blockIdx.y * BN;
  const int tid   = threadIdx.x;
  const int wave  = tid >> 5;
  const int lane  = tid & 31;

  const int cnt = counts[e];                       // uniform scalar load
  if (tileM * TILE_M >= cnt) return;               // uniform early exit
  const int nvalid = min(TILE_M, cnt - tileM * TILE_M);

  if (tid < TILE_M) {
    int g = tileM * TILE_M + tid;
    tok[tid] = (g < cnt) ? idx[e * T_TOKENS + g] : 0;
  }

  // B tile base (transposed weights): WeT[e][n0blk + n][k], row stride C.
  const size_t bbase = (size_t)(uintptr_t)WeT +
                       ((size_t)e * D_HIDDEN * D_MODEL + (size_t)n0blk * D_MODEL) * 2;

  if (wave == 0) {
    // ---- A: gather-mode TDM load, 16-bit row indices = token ids ----
    {
      int base = e * T_TOKENS + tileM * TILE_M;
      unsigned p[8];
      for (int j = 0; j < 8; ++j) {
        unsigned lo = (2*j     < nvalid) ? (unsigned)idx[base + 2*j]     & 0xFFFFu : 0u;
        unsigned hi = (2*j + 1 < nvalid) ? (unsigned)idx[base + 2*j + 1] & 0xFFFFu : 0u;
        p[j] = lo | (hi << 16);
      }
      size_t ga = (size_t)(uintptr_t)xbf;
      u32x4 g0 = { 1u | (1u << 31),                       // count=1, gather, 16-bit idx
                   lds_off(&Abuf[0]),
                   (unsigned)ga,
                   (unsigned)(ga >> 32) | (2u << 30) };   // type=2
      i32x8 g1 = { (int)((1u << 16) | (1u << 20) | (7u << 22) | (3u << 25)),
                   //  data_size=2B | pad_en | interval: 1KB | amount: 16B
                   (int)((D_MODEL & 0xFFFF) << 16),       // tensor_dim0 = 1024
                   (int)((T_TOKENS & 0xFFFFu) << 16),     // tensor_dim1 = 16384 rows
                   (int)((D_MODEL & 0xFFFF) << 16),       // tile_dim0 = 1024 (row width)
                   nvalid,                                // tile_dim1 = #valid indices
                   D_MODEL,                               // tensor_dim0_stride = 1024
                   0, 0 };
      i32x4 g2 = { (int)p[0], (int)p[1], (int)p[2], (int)p[3] };
      i32x4 g3 = { (int)p[4], (int)p[5], (int)p[6], (int)p[7] };
      tdm_issue(g0, g1, g2, g3);
    }
    // ---- B: first 2D tile (128 x 64 bf16) ----
    {
      size_t ga = bbase;                                   // k offset 0
      u32x4 g0 = { 1u, lds_off(&Bbuf[0][0]),
                   (unsigned)ga, (unsigned)(ga >> 32) | (2u << 30) };
      i32x8 g1 = { (int)((1u << 16) | (1u << 20) | (4u << 22) | (7u << 25)),
                   //  data_size=2B | pad_en | interval: 128B | amount: 32B
                   (int)((D_MODEL & 0xFFFF) << 16),        // tensor_dim0 = 1024 (k)
                   (int)((D_HIDDEN & 0xFFFF) << 16),       // tensor_dim1 = 4096 (n)
                   (int)(KSTEP << 16),                     // tile_dim0 = 64
                   BN,                                     // tile_dim1 = 128
                   D_MODEL,                                // dim0 stride = 1024
                   0, 0 };
      i32x4 gz = { 0, 0, 0, 0 };
      tdm_issue(g0, g1, gz, gz);
    }
  }

  const int nW    = wave * 16;
  const int mRow  = lane & 15;
  const int hiSel = lane >> 4;
  v8f acc = {};

  const char* Arow = &Abuf[mRow * A_ROW_BYTES];
  const char* Brow0 = &Bbuf[0][(nW + mRow) * B_ROW_BYTES + hiSel * 16];
  const char* Brow1 = &Bbuf[1][(nW + mRow) * B_ROW_BYTES + hiSel * 16];

  for (int it = 0; it < NITER; ++it) {
    if (wave == 0) {
      if (it + 1 < NITER) {                         // prefetch next B tile
        size_t ga = bbase + (size_t)(it + 1) * KSTEP * 2;
        u32x4 g0 = { 1u, lds_off(&Bbuf[(it + 1) & 1][0]),
                     (unsigned)ga, (unsigned)(ga >> 32) | (2u << 30) };
        i32x8 g1 = { (int)((1u << 16) | (1u << 20) | (4u << 22) | (7u << 25)),
                     (int)((D_MODEL & 0xFFFF) << 16),
                     (int)((D_HIDDEN & 0xFFFF) << 16),
                     (int)(KSTEP << 16),
                     BN,
                     D_MODEL,
                     0, 0 };
        i32x4 gz = { 0, 0, 0, 0 };
        tdm_issue(g0, g1, gz, gz);
        __builtin_amdgcn_s_wait_tensorcnt(1);       // current tile (+A) done
      } else {
        __builtin_amdgcn_s_wait_tensorcnt(0);       // last tile done
      }
    }
    __syncthreads();                                // staged data visible

    // ---- fragments ----
    // A (16x32 bf16 per WMMA): lane = row mRow; K = kbase + hiSel*8 + {0..7},
    // and + 16 for the upper VGPR half (ISA 16-bit A layout). LDS offset
    // includes the TDM pad: +16B per 1KB chunk.
    const int kb = it * KSTEP;
    const int k0 = kb + hiSel * 8;
    const int k1 = kb + 32 + hiSel * 8;             // second WMMA of this iter
    v8bf a0lo = *(const v8bf*)(Arow + k0 * 2        + ((k0      ) >> 9) * 16);
    v8bf a0hi = *(const v8bf*)(Arow + (k0 + 16) * 2 + ((k0 + 16) >> 9) * 16);
    v8bf a1lo = *(const v8bf*)(Arow + k1 * 2        + ((k1      ) >> 9) * 16);
    v8bf a1hi = *(const v8bf*)(Arow + (k1 + 16) * 2 + ((k1 + 16) >> 9) * 16);

    // B (32x16 bf16 per WMMA): lane = col mRow; K = kbase + hiSel*8 + {0..7},
    // and + 16 for the upper half (same as A). Row stride 160B (TDM pad). WMMA0: tile-k 0..31, WMMA1: 32..63.
    const char* Br = (it & 1) ? Brow1 : Brow0;
    v8bf b0a = *(const v8bf*)(Br);
    v8bf b0b = *(const v8bf*)(Br + 32);
    v8bf b1a = *(const v8bf*)(Br + 64);
    v8bf b1b = *(const v8bf*)(Br + 96);

    v16bf a, b;
    for (int j = 0; j < 8; ++j) { a[j] = a0lo[j]; a[8 + j] = a0hi[j];
                                  b[j] = b0a[j];  b[8 + j] = b0b[j]; }
    acc = __builtin_amdgcn_wmma_f32_16x16x32_bf16(false, a, false, b,
                                                  (short)0, acc, false, false);
    for (int j = 0; j < 8; ++j) { a[j] = a1lo[j]; a[8 + j] = a1hi[j];
                                  b[j] = b1a[j];  b[8 + j] = b1b[j]; }
    acc = __builtin_amdgcn_wmma_f32_16x16x32_bf16(false, a, false, b,
                                                  (short)0, acc, false, false);

    __syncthreads();                                // reads done before rewrite
  }

  // Epilogue: C/D layout -> VGPR r holds row M = r + 8*hiSel, col N = lane&15.
  {
    int gn = n0blk + nW + mRow;
    float bias = be[(size_t)e * D_HIDDEN + gn];
    for (int r = 0; r < 8; ++r) {
      int M = r + 8 * hiSel;
      if (tileM * TILE_M + M < cnt)
        out[(size_t)tok[M] * D_HIDDEN + gn] = acc[r] + bias;
    }
  }
}

// ---------------------------------------------------------------------------
extern "C" void kernel_launch(void* const* d_in, const int* in_sizes, int n_in,
                              void* d_out, int out_size, void* d_ws, size_t ws_size,
                              hipStream_t stream) {
  (void)in_sizes; (void)n_in; (void)out_size; (void)ws_size;
  const float* x  = (const float*)d_in[0];
  const float* Wg = (const float*)d_in[1];
  const float* bg = (const float*)d_in[2];
  const float* We = (const float*)d_in[3];
  const float* be = (const float*)d_in[4];
  float* out = (float*)d_out;

  char* ws = (char*)d_ws;
  int*    counts = (int*)ws;
  int*    idx    = (int*)(ws + WS_IDX_OFF);
  __bf16* xbf    = (__bf16*)(ws + WS_XBF_OFF);
  __bf16* WeT    = (__bf16*)(ws + WS_WET_OFF);

  moe_init<<<1, 32, 0, stream>>>(counts);
  moe_cvt_x<<<(T_TOKENS * (size_t)D_MODEL) / (256 * 8), 256, 0, stream>>>(x, xbf);
  moe_cvt_we<<<dim3(D_MODEL / 32, D_HIDDEN / 32, NUM_EXPERT), 256, 0, stream>>>(We, WeT);
  moe_gate<<<T_TOKENS / 256, 256, 0, stream>>>(x, Wg, bg, counts, idx);

  dim3 grid(NUM_EXPERT * MTILES_PER_EXPERT, D_HIDDEN / BN);   // early-exit on count
  moe_expert_gemm<<<grid, 256, 0, stream>>>(xbf, WeT, be, counts, idx, out);
}
